// gru1_39376260170353
// MI455X (gfx1250) — matrix operand order repaired and checked by reading, not compiled, not run on hardware
//
#include <hip/hip_runtime.h>
#include <math.h>

typedef __attribute__((ext_vector_type(16))) __bf16 v16bf;
typedef __attribute__((ext_vector_type(8)))  float  v8f;
typedef __attribute__((ext_vector_type(4)))  unsigned int uint4v;

#define NI 1024
#define NH 1024
#define BB 64
#define TT 512

#define PIPE_D 4                 // async pipeline depth (stages in flight)
#define STAGE_BYTES 4096         // per-wave stage: 4 n-tiles * 32 lanes * 32 B

// ---------- helpers ----------

__device__ __forceinline__ unsigned short f2bf(float x) {
  union { float f; unsigned int u; } v; v.f = x;
  unsigned int r = v.u + 0x7FFFu + ((v.u >> 16) & 1u); // round-to-nearest-even
  return (unsigned short)(r >> 16);
}

union AFrag { uint4v q[2]; v16bf v; };
union BFrag { uint4v q[2]; v16bf v; };

// A fragment: 16x32 bf16 (MxK), ISA 7.12.2. For lane = half*16 + m the 16
// elements are one contiguous 16-element run (k = 16*half + s; this code's order, not the ISA table's) -> two 128-bit loads.
__device__ __forceinline__ void load_a_frag(AFrag& a, const unsigned short* __restrict__ A,
                                            int lda, int m, int half, int kBase) {
  const unsigned short* row = A + (size_t)m * lda + kBase + half * 16;
  a.q[0] = *(const uint4v*)(row);
  a.q[1] = *(const uint4v*)(row + 8);
}

// B fragment from packed weights: layout [K/32][N/16][lane(32)][j(16)] bf16,
// each lane's fragment is one contiguous 32-byte chunk.
__device__ __forceinline__ void load_b_frag(BFrag& b, const unsigned short* __restrict__ Wp,
                                            int ntiles, int kt, int nt, int lane) {
  const unsigned short* p = Wp + ((((size_t)kt * ntiles + nt) * 32 + lane) << 4);
  b.v = *(const v16bf*)(p);
}

__device__ __forceinline__ float sigmoidf_(float x) { return 1.0f / (1.0f + __expf(-x)); }

// Issue one pipeline stage: async-copy this wave's 4 B fragments (4 KB) for
// K-tile `kt` from packed weights into this wave's LDS ring slot.
// 8 GLOBAL_LOAD_ASYNC_TO_LDS_B128 per lane -> ASYNCcnt += 8 for the wave.
__device__ __forceinline__ void issue_stage_async(const unsigned short* __restrict__ Wp,
                                                  int ntiles, int kt, int nt0, int lane,
                                                  unsigned char* slotPtr) {
#pragma unroll
  for (int nb = 0; nb < 4; ++nb) {
    const unsigned short* src =
        Wp + ((((size_t)kt * ntiles + (nt0 + nb)) * 32 + lane) << 4);
    // generic pointer low 32 bits == LDS offset
    unsigned dst = (unsigned)(size_t)(slotPtr + nb * 1024 + lane * 32);
    asm volatile("global_load_async_to_lds_b128 %0, %1, off"
                 :: "v"(dst), "v"(src) : "memory");
    asm volatile("global_load_async_to_lds_b128 %0, %1, off"
                 :: "v"(dst + 16u), "v"((const unsigned short*)((const char*)src + 16))
                 : "memory");
  }
}

// ---- register-pipelined strip GEMM (used by high-occupancy xproj phase) ----
__device__ __forceinline__ void gemm_strip(v8f (&acc)[4],
                                           const unsigned short* __restrict__ A, int lda,
                                           const unsigned short* __restrict__ Wp,
                                           int ntiles, int KT,
                                           int m, int half, int nt0, int lane) {
  AFrag a[2];
  BFrag b[2][4];
  load_a_frag(a[0], A, lda, m, half, 0);
#pragma unroll
  for (int nb = 0; nb < 4; ++nb) load_b_frag(b[0][nb], Wp, ntiles, 0, nt0 + nb, lane);

#pragma unroll 2
  for (int kt = 0; kt < KT - 1; ++kt) {
    const int cur = kt & 1;
    const int nxt = cur ^ 1;
    if (kt + 2 < KT)
      __builtin_prefetch(Wp + ((((size_t)(kt + 2) * ntiles + nt0) * 32 + lane) << 4), 0, 1);
    load_a_frag(a[nxt], A, lda, m, half, (kt + 1) << 5);
#pragma unroll
    for (int nb = 0; nb < 4; ++nb)
      load_b_frag(b[nxt][nb], Wp, ntiles, kt + 1, nt0 + nb, lane);
#pragma unroll
    for (int nb = 0; nb < 4; ++nb)
      acc[nb] = __builtin_amdgcn_wmma_f32_16x16x32_bf16(
          false, a[cur].v, false, b[cur][nb].v, (short)0, acc[nb], false, false);
  }
  {
    const int cur = (KT - 1) & 1;
#pragma unroll
    for (int nb = 0; nb < 4; ++nb)
      acc[nb] = __builtin_amdgcn_wmma_f32_16x16x32_bf16(
          false, a[cur].v, false, b[cur][nb].v, (short)0, acc[nb], false, false);
  }
}

// ---- async-LDS pipelined strip GEMM (latency-bound recurrent phase) ----
// Per-wave private LDS ring of PIPE_D stages. Weights stream via
// GLOBAL_LOAD_ASYNC_TO_LDS_B128 (ASYNCcnt) with PIPE_D stages in flight;
// consumption reads LDS (ds_load_b128, short latency). No barriers needed:
// each wave only touches its own ring slots.
__device__ __forceinline__ void gemm_strip_async(v8f (&acc)[4],
                                                 const unsigned short* __restrict__ A, int lda,
                                                 const unsigned short* __restrict__ Wp,
                                                 int ntiles, int KT,
                                                 int m, int half, int nt0, int lane,
                                                 unsigned char* ring) {
  // prologue: fill the pipeline
#pragma unroll
  for (int s = 0; s < PIPE_D; ++s)
    issue_stage_async(Wp, ntiles, s, nt0, lane, ring + s * STAGE_BYTES);

  AFrag a[2];
  load_a_frag(a[0], A, lda, m, half, 0);

#pragma unroll 4
  for (int kt = 0; kt < KT; ++kt) {
    const int slot = kt & (PIPE_D - 1);
    const int cur  = kt & 1;
    if (kt + 1 < KT)                       // overlap next A load with the wait
      load_a_frag(a[cur ^ 1], A, lda, m, half, (kt + 1) << 5);
    // stage kt complete when <= (PIPE_D-1)*8 = 24 async ops outstanding
    asm volatile("s_wait_asynccnt 0x18" ::: "memory");
    BFrag b[4];
    unsigned char* sp = ring + slot * STAGE_BYTES + lane * 32;
#pragma unroll
    for (int nb = 0; nb < 4; ++nb)
      b[nb].v = *(const v16bf*)(sp + nb * 1024);
#pragma unroll
    for (int nb = 0; nb < 4; ++nb)
      acc[nb] = __builtin_amdgcn_wmma_f32_16x16x32_bf16(
          false, a[cur].v, false, b[nb].v, (short)0, acc[nb], false, false);
    if (kt + PIPE_D < KT) {
      // our LDS reads of this slot are done before the async engine refills it
      asm volatile("s_wait_dscnt 0x0" ::: "memory");
      issue_stage_async(Wp, ntiles, kt + PIPE_D, nt0, lane, ring + slot * STAGE_BYTES);
    }
  }
  asm volatile("s_wait_asynccnt 0x0" ::: "memory");  // quiesce before return
}

// ---------- utility kernels ----------

__global__ void cvt_f32_to_bf16(const float* __restrict__ src,
                                unsigned short* __restrict__ dst, int n) {
  int i = blockIdx.x * blockDim.x + threadIdx.x;
  int stride = gridDim.x * blockDim.x;
  for (; i < n; i += stride) dst[i] = f2bf(src[i]);
}

// Convert f32 [K,N] weight to bf16 packed fragment-major layout:
// dst[ ((kt*(N/16)+nt)*32 + (half*16+nn)) * 16 + j ] = src[k*N+n]
__global__ void pack_w_bf16(const float* __restrict__ src,
                            unsigned short* __restrict__ dst, int K, int N) {
  int i = blockIdx.x * blockDim.x + threadIdx.x;
  int stride = gridDim.x * blockDim.x;
  int total = K * N;
  for (; i < total; i += stride) {
    int k = i / N, n = i - k * N;
    int kt = k >> 5, kr = k & 31;
    int half = kr >> 4, j = kr & 15;
    int nt = n >> 4, nn = n & 15;
    int lane = half * 16 + nn;
    size_t di = ((((size_t)kt * (N >> 4) + nt) * 32 + lane) << 4) + j;
    dst[di] = f2bf(src[i]);
  }
}

__global__ void fill_f32(float* __restrict__ p, float v, int n) {
  int i = blockIdx.x * blockDim.x + threadIdx.x;
  int stride = gridDim.x * blockDim.x;
  for (; i < n; i += stride) p[i] = v;
}

__global__ void fill_u16(unsigned short* __restrict__ p, unsigned short v, int n) {
  int i = blockIdx.x * blockDim.x + threadIdx.x;
  int stride = gridDim.x * blockDim.x;
  for (; i < n; i += stride) p[i] = v;
}

// ---------- phase 1: input projections  X = in @ W + b ----------
__global__ void __launch_bounds__(256)
xproj_gemm(const unsigned short* __restrict__ Abf,  // [M,K] bf16 row-major
           const unsigned short* __restrict__ Wz,   // packed
           const unsigned short* __restrict__ Wr,
           const unsigned short* __restrict__ Wh,
           const float* __restrict__ bz, const float* __restrict__ br,
           const float* __restrict__ bh,
           float* __restrict__ Xz, float* __restrict__ Xr, float* __restrict__ Xh) {
  const int lane  = threadIdx.x & 31;
  const int wave  = threadIdx.x >> 5;
  const int m0    = blockIdx.y * 128 + wave * 16;
  const int nt0   = blockIdx.x * 4;
  const int gate  = blockIdx.z;
  const unsigned short* W = (gate == 0) ? Wz : ((gate == 1) ? Wr : Wh);
  const float* bias       = (gate == 0) ? bz : ((gate == 1) ? br : bh);
  float* X                = (gate == 0) ? Xz : ((gate == 1) ? Xr : Xh);

  const int mfrag = lane & 15;
  const int half  = lane >> 4;

  v8f acc[4] = {};
  gemm_strip(acc, Abf, NI, W, NH >> 4, NI >> 5, m0 + mfrag, half, nt0, lane);

#pragma unroll
  for (int nb = 0; nb < 4; ++nb) {
    int col = ((nt0 + nb) << 4) + mfrag;
    float bv = bias[col];
#pragma unroll
    for (int e = 0; e < 8; ++e) {
      int row = m0 + half * 8 + e;
      X[(size_t)row * NH + col] = acc[nb][e] + bv;
    }
  }
}

// ---------- phase 2a: Z and R gates for one time step ----------
__global__ void __launch_bounds__(128)
step_zr(const unsigned short* __restrict__ Hb,   // [B,NH] bf16
        const float* __restrict__ H,             // [B,NH] f32
        const unsigned short* __restrict__ Whz,  // packed
        const unsigned short* __restrict__ Whr,  // packed
        const float* __restrict__ Xz,            // [B,T,NH]
        const float* __restrict__ Xr,
        float* __restrict__ Z,                   // [B,NH]
        unsigned short* __restrict__ RHb,        // [B,NH] bf16  (R .* H)
        int t) {
  __shared__ unsigned char smem[4 * PIPE_D * STAGE_BYTES];  // 64 KB: 4 waves
  const int lane  = threadIdx.x & 31;
  const int wave  = threadIdx.x >> 5;
  const int m0    = wave * 16;
  const int nt0   = blockIdx.x * 4;
  const int gate  = blockIdx.z;   // 0=z, 1=r
  const unsigned short* W = (gate == 0) ? Whz : Whr;
  const float* Xg         = (gate == 0) ? Xz : Xr;

  const int mfrag = lane & 15;
  const int half  = lane >> 4;
  unsigned char* ring = &smem[wave * PIPE_D * STAGE_BYTES];

  v8f acc[4] = {};
  gemm_strip_async(acc, Hb, NH, W, NH >> 4, NH >> 5, m0 + mfrag, half, nt0, lane, ring);

#pragma unroll
  for (int nb = 0; nb < 4; ++nb) {
    int col = ((nt0 + nb) << 4) + mfrag;
#pragma unroll
    for (int e = 0; e < 8; ++e) {
      int b = m0 + half * 8 + e;  // batch row
      float x = Xg[((size_t)b * TT + t) * NH + col];
      float g = sigmoidf_(x + acc[nb][e]);
      size_t idx = (size_t)b * NH + col;
      if (gate == 0) {
        Z[idx] = g;
      } else {
        RHb[idx] = f2bf(g * H[idx]);
      }
    }
  }
}

// ---------- phase 2b: candidate + state update for one time step ----------
__global__ void __launch_bounds__(128)
step_h(const unsigned short* __restrict__ RHb,   // [B,NH] bf16
       const unsigned short* __restrict__ Whh,   // packed
       const float* __restrict__ Xh,             // [B,T,NH]
       const float* __restrict__ Z,              // [B,NH]
       float* __restrict__ H,                    // [B,NH] same-index in-place update
       unsigned short* __restrict__ Hb,          // [B,NH] bf16 mirror of H
       float* __restrict__ out,                  // [B,T,NH]
       float* __restrict__ hT,                   // final-state slot or nullptr
       int t) {
  __shared__ unsigned char smem[4 * PIPE_D * STAGE_BYTES];  // 64 KB: 4 waves
  const int lane  = threadIdx.x & 31;
  const int wave  = threadIdx.x >> 5;
  const int m0    = wave * 16;
  const int nt0   = blockIdx.x * 4;

  const int mfrag = lane & 15;
  const int half  = lane >> 4;
  unsigned char* ring = &smem[wave * PIPE_D * STAGE_BYTES];

  v8f acc[4] = {};
  gemm_strip_async(acc, RHb, NH, Whh, NH >> 4, NH >> 5, m0 + mfrag, half, nt0, lane, ring);

#pragma unroll
  for (int nb = 0; nb < 4; ++nb) {
    int col = ((nt0 + nb) << 4) + mfrag;
#pragma unroll
    for (int e = 0; e < 8; ++e) {
      int b = m0 + half * 8 + e;  // batch row
      size_t idx = (size_t)b * NH + col;
      float htilde = tanhf(Xh[((size_t)b * TT + t) * NH + col] + acc[nb][e]);
      float z = Z[idx];
      float hn = z * H[idx] + (1.0f - z) * htilde;
      H[idx]  = hn;
      Hb[idx] = f2bf(hn);
      out[((size_t)b * TT + t) * NH + col] = hn;
      if (hT) hT[idx] = hn;
    }
  }
}

// ---------- host ----------

extern "C" void kernel_launch(void* const* d_in, const int* in_sizes, int n_in,
                              void* d_out, int out_size, void* d_ws, size_t ws_size,
                              hipStream_t stream) {
  const float* inputs = (const float*)d_in[0];
  const float* W_xz   = (const float*)d_in[1];
  const float* W_hz   = (const float*)d_in[2];
  const float* b_z    = (const float*)d_in[3];
  const float* W_xr   = (const float*)d_in[4];
  const float* W_hr   = (const float*)d_in[5];
  const float* b_r    = (const float*)d_in[6];
  const float* W_xh   = (const float*)d_in[7];
  const float* W_hh   = (const float*)d_in[8];
  const float* b_h    = (const float*)d_in[9];
  float* out = (float*)d_out;  // [B,T,NH] outputs then [B,NH] H_T

  const size_t M = (size_t)BB * TT;  // 32768

  // workspace layout
  char* ws = (char*)d_ws;
  float* Xz = (float*)ws; ws += M * NH * sizeof(float);
  float* Xr = (float*)ws; ws += M * NH * sizeof(float);
  float* Xh = (float*)ws; ws += M * NH * sizeof(float);
  unsigned short* inb  = (unsigned short*)ws; ws += M * NI * sizeof(unsigned short);
  unsigned short* wxzb = (unsigned short*)ws; ws += (size_t)NI * NH * sizeof(unsigned short);
  unsigned short* whzb = (unsigned short*)ws; ws += (size_t)NH * NH * sizeof(unsigned short);
  unsigned short* wxrb = (unsigned short*)ws; ws += (size_t)NI * NH * sizeof(unsigned short);
  unsigned short* whrb = (unsigned short*)ws; ws += (size_t)NH * NH * sizeof(unsigned short);
  unsigned short* wxhb = (unsigned short*)ws; ws += (size_t)NI * NH * sizeof(unsigned short);
  unsigned short* whhb = (unsigned short*)ws; ws += (size_t)NH * NH * sizeof(unsigned short);
  float* H  = (float*)ws;           ws += (size_t)BB * NH * sizeof(float);
  float* Zb = (float*)ws;           ws += (size_t)BB * NH * sizeof(float);
  unsigned short* Hb  = (unsigned short*)ws; ws += (size_t)BB * NH * sizeof(unsigned short);
  unsigned short* RHb = (unsigned short*)ws; ws += (size_t)BB * NH * sizeof(unsigned short);

  // conversions: activations row-major bf16, weights packed fragment-major bf16
  {
    int n = (int)(M * NI);
    cvt_f32_to_bf16<<<4096, 256, 0, stream>>>(inputs, inb, n);
    pack_w_bf16<<<1024, 256, 0, stream>>>(W_xz, wxzb, NI, NH);
    pack_w_bf16<<<1024, 256, 0, stream>>>(W_hz, whzb, NH, NH);
    pack_w_bf16<<<1024, 256, 0, stream>>>(W_xr, wxrb, NI, NH);
    pack_w_bf16<<<1024, 256, 0, stream>>>(W_hr, whrb, NH, NH);
    pack_w_bf16<<<1024, 256, 0, stream>>>(W_xh, wxhb, NI, NH);
    pack_w_bf16<<<1024, 256, 0, stream>>>(W_hh, whhb, NH, NH);
  }
  // H0 = 0
  fill_f32<<<256, 256, 0, stream>>>(H, 0.0f, BB * NH);
  fill_u16<<<256, 256, 0, stream>>>(Hb, (unsigned short)0, BB * NH);

  // phase 1: input projections (3 gates in one launch via blockIdx.z)
  {
    dim3 grid(NH / 64, (unsigned)(M / 128), 3);
    xproj_gemm<<<grid, 256, 0, stream>>>(inb, wxzb, wxrb, wxhb, b_z, b_r, b_h,
                                         Xz, Xr, Xh);
  }

  // phase 2: sequential recurrence, 2 kernels per step on the same stream
  float* hT = out + M * NH;
  for (int t = 0; t < TT; ++t) {
    dim3 gzr(NH / 64, 1, 2);
    step_zr<<<gzr, 128, 0, stream>>>(Hb, H, whzb, whrb, Xz, Xr, Zb, RHb, t);
    dim3 gh(NH / 64, 1, 1);
    step_h<<<gh, 128, 0, stream>>>(RHb, whhb, Xh, Zb, H, Hb, out,
                                   (t == TT - 1) ? hT : (float*)nullptr, t);
  }
}
